// SphericalSpectralConv2D_38405597561164
// MI455X (gfx1250) — matrix operand order repaired and checked by reading, not compiled, not run on hardware
//
#include <hip/hip_runtime.h>
#include <hip/hip_bf16.h>

#define LMAX   128
#define NTH    256
#define NPH    512
#define NMODES 255
#define CIN    32
#define COUT   32

typedef __attribute__((ext_vector_type(16))) __bf16 v16bf;
typedef __attribute__((ext_vector_type(8)))  __bf16 v8bf;
typedef __attribute__((ext_vector_type(8)))  float  v8f;

// ---- TDM availability / arity detection ----
#if defined(__AMDGCN__) && __has_builtin(__builtin_amdgcn_tensor_load_to_lds) && \
    __has_builtin(__builtin_amdgcn_s_wait_tensorcnt)
#define USE_TDM 1
#else
#define USE_TDM 0
#endif
#if __has_include(<hip/amd_detail/amd_gfx1250_TDM.h>)
#define TDM_SIX_ARGS 1
#else
#define TDM_SIX_ARGS 0
#endif

#if USE_TDM
typedef unsigned int tdm_v4u __attribute__((ext_vector_type(4)));
typedef int          tdm_v8i __attribute__((ext_vector_type(8)));
typedef int          tdm_v4i __attribute__((ext_vector_type(4)));

// 2D tensor-DMA: copy tile_h rows of tile_w bf16 elems (global row stride
// row_stride elems) into LDS with hardware padding (pad codes per D# spec).
static __device__ __forceinline__ void tdm_load_2d(const void* gsrc, void* ldst,
                                                   unsigned tile_w, unsigned tile_h,
                                                   unsigned row_stride, unsigned tensor_w,
                                                   unsigned pad_int_code, unsigned pad_amt_code) {
  unsigned long long ga = (unsigned long long)(uintptr_t)gsrc;
  unsigned la = (unsigned)(uintptr_t)ldst;            // low 32 bits = LDS byte address
  tdm_v4u g0 = { 1u,                                   // count=1, is_restore=0, gather=0
                 la,                                   // lds_addr
                 (unsigned)ga,                         // global_addr[31:0]
                 (unsigned)((ga >> 32) & 0x01FFFFFFu) | (2u << 30) }; // [56:32] | type=2
  tdm_v8i g1;
  g1[0] = (int)((1u << 16) |                           // data_size = 2 bytes
                (1u << 20) |                           // pad_enable
                (pad_int_code << 22) | (pad_amt_code << 25));
  g1[1] = (int)((tensor_w & 0xFFFFu) << 16);           // tensor_dim0[15:0] (atomic_barrier=0)
  g1[2] = (int)(((tensor_w >> 16) & 0xFFFFu) | ((tile_h & 0xFFFFu) << 16)); // tensor_dim1=tile_h
  g1[3] = (int)(((tile_h >> 16) & 0xFFFFu) | ((tile_w & 0xFFFFu) << 16));   // tile_dim0
  g1[4] = (int)(tile_h & 0xFFFFu);                     // tile_dim1 (tile_dim2=0)
  g1[5] = (int)row_stride;                             // tensor_dim0_stride[31:0]
  g1[6] = 0;
  g1[7] = 0;
  tdm_v4i z4 = { 0, 0, 0, 0 };
#if TDM_SIX_ARGS
  tdm_v8i z8 = {};
  __builtin_amdgcn_tensor_load_to_lds(g0, g1, z4, z4, z8, 0);
#else
  __builtin_amdgcn_tensor_load_to_lds(g0, g1, z4, z4, 0);
#endif
}
#endif // USE_TDM

// ---- WMMA fragment helpers (CDNA5 wave32 16x16x32 bf16 layouts) ----
static __device__ __forceinline__ v16bf frag_a(const __bf16* base, int ldk, int lane) {
  const __bf16* p = base + (lane & 15) * ldk + ((lane >> 4) << 3);
  v8bf lo = *(const v8bf*)(p);
  v8bf hi = *(const v8bf*)(p + 16);
  return __builtin_shufflevector(lo, hi, 0,1,2,3,4,5,6,7,8,9,10,11,12,13,14,15);
}
static __device__ __forceinline__ v16bf frag_b(const __bf16* base, int ldk, int lane) {
  const __bf16* p = base + (lane & 15) * ldk + ((lane >> 4) << 3);
  v8bf lo = *(const v8bf*)(p);
  v8bf hi = *(const v8bf*)(p + 16);
  return __builtin_shufflevector(lo, hi, 0,1,2,3,4,5,6,7,8,9,10,11,12,13,14,15);
}
static __device__ __forceinline__ v8f wmma_bf16(v16bf a, v16bf b, v8f c) {
  return __builtin_amdgcn_wmma_f32_16x16x32_bf16(false, a, false, b, (short)0, c, false, false);
}
static __device__ __forceinline__ void store_tile(float* base, int ldc, int lane, v8f c) {
  int nn = lane & 15;
  int m0 = (lane >> 4) << 3;
  #pragma unroll
  for (int r = 0; r < 8; ++r) base[(m0 + r) * ldc + nn] = c[r];
}

// ---- bf16 trig tables: ATF[256][512] fwd-DFT (cos*s | -sin*s),
//      ATI[512][256] inv-DFT (cos | -sin). s = 2*pi/NPH ----
__global__ void k_tables(__bf16* __restrict__ atf, __bf16* __restrict__ ati) {
  int g = blockIdx.x * blockDim.x + threadIdx.x;
  const float s = 6.28318530717958647692f / (float)NPH;
  if (g < 2 * LMAX * NPH) {
    int r = g >> 9, p = g & (NPH - 1);
    int mp = ((r & 127) * p) & (NPH - 1);
    float th = s * (float)mp;
    float v = (r < LMAX) ? __cosf(th) : -__sinf(th);
    atf[g] = (__bf16)(v * s);
  } else if (g < 4 * LMAX * NPH) {
    int g2 = g - 2 * LMAX * NPH;
    int p = g2 >> 8, k = g2 & 255;
    int mp = ((k & 127) * p) & (NPH - 1);
    float th = s * (float)mp;
    ati[g2] = (__bf16)((k < LMAX) ? __cosf(th) : -__sinf(th));
  }
}

// ---- bf16 repack of legendre: legA[mi][l][t] = leg*qw (kernel-B A operand),
//      legT[mi][t][l] = leg (kernel-D A operand, pre-transposed). ----
__global__ __launch_bounds__(256) void k_prep(const float* __restrict__ leg,
                                              const float* __restrict__ qw,
                                              __bf16* __restrict__ legA,
                                              __bf16* __restrict__ legT) {
  int g = blockIdx.x * 256 + threadIdx.x;             // over 128*255*256
  int t = g & 255;
  int r = g >> 8;                                     // l*255 + mi
  int mi = r % NMODES, l = r / NMODES;
  float v = leg[g];
  legA[(size_t)mi * (LMAX * NTH) + l * NTH + t] = (__bf16)(v * qw[t]);
  legT[(size_t)mi * (LMAX * NTH) + t * LMAX + l] = (__bf16)v;
}

// ---- Kernel A: forward DFT as matmul. Per theta t:
//      [Fre;Fim](256x32) = ATF(256x512) * x_t(512x32) ----
#define LDBX 528
__global__ __launch_bounds__(256) void k_dft(const float* __restrict__ x,
                                             const __bf16* __restrict__ atf,
                                             float* __restrict__ Fre,
                                             float* __restrict__ Fim) {
  __shared__ alignas(16) __bf16 Bt[32 * LDBX];
  int t = blockIdx.x, tid = threadIdx.x;
  const float* xt = x + t * (NPH * CIN);
  for (int i = 0; i < 64; ++i) {
    int idx = i * 256 + tid;
    int p = idx >> 5, c = idx & 31;
    Bt[c * LDBX + p] = (__bf16)xt[idx];
  }
  __syncthreads();
  int lane = tid & 31, wave = tid >> 5;
  for (int i = 0; i < 4; ++i) {                       // 32 tiles = 16 Mt x 2 Nt
    int tile = wave * 4 + i;
    int mt = tile >> 1, nt = tile & 1;
    v8f acc = {};
    const __bf16* ab = atf + (mt * 16) * NPH;
    const __bf16* bb = Bt + (nt * 16) * LDBX;
    #pragma unroll
    for (int ks = 0; ks < 16; ++ks)
      acc = wmma_bf16(frag_a(ab + ks * 32, NPH, lane),
                      frag_b(bb + ks * 32, LDBX, lane), acc);
    float* dst = (mt < 8 ? Fre : Fim) + t * (LMAX * CIN) + ((mt & 7) * 16) * CIN + nt * 16;
    store_tile(dst, CIN, lane, acc);
  }
}

// ---- Kernel B: forward Legendre. Per mi:
//      flm(128 x [Re32|Im32]) = legA_mi(128x256) * Fsel^T(256x64). K chunks of 64.
//      A tile DMA'd by TDM (hardware row padding -> 72-half stride). ----
#define LDAB 72
__global__ __launch_bounds__(256) void k_leg_fwd(const __bf16* __restrict__ legA,
                                                 const float* __restrict__ Fre,
                                                 const float* __restrict__ Fim,
                                                 float* __restrict__ flmRe,
                                                 float* __restrict__ flmIm) {
  __shared__ alignas(16) __bf16 As[128 * LDAB];
  __shared__ alignas(16) __bf16 Bs[64 * LDAB];
  int mi = blockIdx.x, tid = threadIdx.x;
  int m = mi - (LMAX - 1);
  int am = m < 0 ? -m : m;
  float sgn = m < 0 ? -1.f : 1.f;
  int lane = tid & 31, wave = tid >> 5;
  const __bf16* legAm = legA + (size_t)mi * (LMAX * NTH);
  v8f acc[4] = {};
  for (int kc = 0; kc < 4; ++kc) {                    // K = 256 in chunks of 64
    int t0 = kc * 64;
#if USE_TDM
    if (tid == 0)                                     // one DMA per workgroup
      tdm_load_2d(legAm + t0, As, /*tile_w*/64, /*tile_h*/128,
                  /*row_stride*/NTH, /*tensor_w*/64,
                  /*pad: every 32 dwords*/4, /*add 4 dwords*/3);   // 64+8 = 72 halfs
#else
    for (int i = 0; i < 4; ++i) {                     // 1024 chunks of 8 halfs
      int c = i * 256 + tid;
      int l = c >> 3, off = (c & 7) * 8;
      *(v8bf*)&As[l * LDAB + off] = *(const v8bf*)&legAm[l * NTH + t0 + off];
    }
#endif
    for (int i = 0; i < 16; ++i) {                    // B chunk 64x64 (N x K), f32->bf16
      int idx = i * 256 + tid;
      int n = idx & 63, tt = idx >> 6;
      float v = (n < 32) ? Fre[(t0 + tt) * (LMAX * CIN) + am * CIN + n]
                         : sgn * Fim[(t0 + tt) * (LMAX * CIN) + am * CIN + (n - 32)];
      Bs[n * LDAB + tt] = (__bf16)v;
    }
#if USE_TDM
    if (tid == 0) __builtin_amdgcn_s_wait_tensorcnt(0);
#endif
    __syncthreads();
    for (int i = 0; i < 4; ++i) {                     // 32 tiles = 8 Mt x 4 Nt
      int tile = wave * 4 + i;
      int mt = tile >> 2, nt = tile & 3;
      const __bf16* ab = As + (mt * 16) * LDAB;
      const __bf16* bb = Bs + (nt * 16) * LDAB;
      #pragma unroll
      for (int ks = 0; ks < 2; ++ks)
        acc[i] = wmma_bf16(frag_a(ab + ks * 32, LDAB, lane),
                           frag_b(bb + ks * 32, LDAB, lane), acc[i]);
    }
    __syncthreads();
  }
  for (int i = 0; i < 4; ++i) {
    int tile = wave * 4 + i;
    int mt = tile >> 2, nt = tile & 3;
    float* dst = (nt < 2 ? flmRe : flmIm) + (mt * 16) * (NMODES * CIN) + mi * CIN + (nt & 1) * 16;
    store_tile(dst, NMODES * CIN, lane, acc[i]);
  }
}

// ---- Kernel C: per-(l,m) complex 32x32 weight apply (bandwidth bound). ----
__global__ __launch_bounds__(256) void k_wapply(const float* __restrict__ flmRe,
                                                const float* __restrict__ flmIm,
                                                const float* __restrict__ Wr,
                                                const float* __restrict__ Wi,
                                                float* __restrict__ yRe,
                                                float* __restrict__ yIm) {
  int wave = threadIdx.x >> 5, lane = threadIdx.x & 31;
  int pair = blockIdx.x * 8 + wave;                   // 128*255 = 32640 pairs
  int l = pair / NMODES, mi = pair % NMODES;
  const float* fr = flmRe + l * (NMODES * CIN) + mi * CIN;
  const float* fi = flmIm + l * (NMODES * CIN) + mi * CIN;
  const float* wr = Wr + (l * NMODES + mi) * (CIN * COUT) + lane;
  const float* wi = Wi + (l * NMODES + mi) * (CIN * COUT) + lane;
  float ar = 0.f, ai = 0.f;
  #pragma unroll 8
  for (int c = 0; c < CIN; ++c) {
    float r = fr[c], q = fi[c];
    float a = wr[c * COUT], b = wi[c * COUT];
    ar += r * a - q * b;
    ai += r * b + q * a;
  }
  yRe[l * (NMODES * COUT) + mi * COUT + lane] = ar;
  yIm[l * (NMODES * COUT) + mi * COUT + lane] = ai;
}

// ---- Kernel D: inverse Legendre. Per mi:
//      G(256 x [Re32|Im32]) = legT_mi(256x128) * y(128x64). K chunks of 64.
//      A tile DMA'd by TDM from pre-transposed legT. ----
#define LDL 72
__global__ __launch_bounds__(256) void k_leg_inv(const __bf16* __restrict__ legT,
                                                 const float* __restrict__ yRe,
                                                 const float* __restrict__ yIm,
                                                 float* __restrict__ GRe,
                                                 float* __restrict__ GIm) {
  __shared__ alignas(16) __bf16 Ls[256 * LDL];
  __shared__ alignas(16) __bf16 Ys[64 * LDL];
  int mi = blockIdx.x, tid = threadIdx.x;
  int lane = tid & 31, wave = tid >> 5;
  const __bf16* legTm = legT + (size_t)mi * (LMAX * NTH);
  v8f acc[8] = {};
  for (int kc = 0; kc < 2; ++kc) {                    // K = 128 in chunks of 64
    int l0 = kc * 64;
#if USE_TDM
    if (tid == 0)
      tdm_load_2d(legTm + l0, Ls, /*tile_w*/64, /*tile_h*/256,
                  /*row_stride*/LMAX, /*tensor_w*/64, 4, 3);       // 72-half stride
#else
    for (int i = 0; i < 8; ++i) {                     // 2048 chunks of 8 halfs
      int c = i * 256 + tid;
      int t = c >> 3, off = (c & 7) * 8;
      *(v8bf*)&Ls[t * LDL + off] = *(const v8bf*)&legTm[t * LMAX + l0 + off];
    }
#endif
    for (int i = 0; i < 16; ++i) {                    // Ys 64x64 (N x K), f32->bf16
      int idx = i * 256 + tid;
      int n = idx & 63, ll = idx >> 6;
      float v = (n < 32) ? yRe[(l0 + ll) * (NMODES * COUT) + mi * COUT + n]
                         : yIm[(l0 + ll) * (NMODES * COUT) + mi * COUT + (n - 32)];
      Ys[n * LDL + ll] = (__bf16)v;
    }
#if USE_TDM
    if (tid == 0) __builtin_amdgcn_s_wait_tensorcnt(0);
#endif
    __syncthreads();
    for (int i = 0; i < 8; ++i) {                     // 64 tiles = 16 Mt x 4 Nt
      int tile = wave * 8 + i;
      int mt = tile >> 2, nt = tile & 3;
      const __bf16* ab = Ls + (mt * 16) * LDL;
      const __bf16* bb = Ys + (nt * 16) * LDL;
      #pragma unroll
      for (int ks = 0; ks < 2; ++ks)
        acc[i] = wmma_bf16(frag_a(ab + ks * 32, LDL, lane),
                           frag_b(bb + ks * 32, LDL, lane), acc[i]);
    }
    __syncthreads();
  }
  for (int i = 0; i < 8; ++i) {
    int tile = wave * 8 + i;
    int mt = tile >> 2, nt = tile & 3;
    float* dst = (nt < 2 ? GRe : GIm) + mi * (NTH * COUT) + (mt * 16) * COUT + (nt & 1) * 16;
    store_tile(dst, COUT, lane, acc[i]);
  }
}

// ---- Kernel E: inverse DFT as matmul. Per theta t:
//      out_t(512x32) = ATI(512x256) * [CRe(128x32); CIm(128x32)] ----
#define LDB2 264
__global__ __launch_bounds__(256) void k_idft(const float* __restrict__ GRe,
                                              const float* __restrict__ GIm,
                                              const __bf16* __restrict__ ati,
                                              float* __restrict__ out) {
  __shared__ alignas(16) __bf16 Bs[32 * LDB2];
  int t = blockIdx.x, tid = threadIdx.x;
  for (int i = 0; i < 32; ++i) {
    int idx = i * 256 + tid;
    int o = idx & 31, k = idx >> 5;
    float v;
    if (k < LMAX) {
      int m = k;
      v = GRe[(127 + m) * (NTH * COUT) + t * COUT + o];
      if (m > 0) v += GRe[(127 - m) * (NTH * COUT) + t * COUT + o];
    } else {
      int m = k - LMAX;
      v = GIm[(127 + m) * (NTH * COUT) + t * COUT + o];
      if (m > 0) v -= GIm[(127 - m) * (NTH * COUT) + t * COUT + o];
    }
    Bs[o * LDB2 + k] = (__bf16)v;
  }
  __syncthreads();
  int lane = tid & 31, wave = tid >> 5;
  for (int i = 0; i < 8; ++i) {                       // 64 tiles = 32 Mt(p) x 2 Nt(o)
    int tile = wave * 8 + i;
    int mt = tile >> 1, nt = tile & 1;
    v8f acc = {};
    const __bf16* ab = ati + (mt * 16) * 256;
    const __bf16* bb = Bs + (nt * 16) * LDB2;
    #pragma unroll
    for (int ks = 0; ks < 8; ++ks)
      acc = wmma_bf16(frag_a(ab + ks * 32, 256, lane),
                      frag_b(bb + ks * 32, LDB2, lane), acc);
    int nn = lane & 15, m0 = (lane >> 4) << 3;
    int o = nt * 16 + nn;
    float* dst = out + o * (NTH * NPH) + t * NPH + mt * 16 + m0;
    #pragma unroll
    for (int r = 0; r < 8; ++r) dst[r] = acc[r];
  }
}

extern "C" void kernel_launch(void* const* d_in, const int* in_sizes, int n_in,
                              void* d_out, int out_size, void* d_ws, size_t ws_size,
                              hipStream_t stream) {
  (void)in_sizes; (void)n_in; (void)out_size; (void)ws_size;
  const float* x   = (const float*)d_in[0];
  const float* Wr  = (const float*)d_in[1];
  const float* Wi  = (const float*)d_in[2];
  const float* leg = (const float*)d_in[3];
  const float* qw  = (const float*)d_in[4];
  float* out = (float*)d_out;

  char* ws = (char*)d_ws;
  size_t off = 0;
  auto take = [&](size_t bytes) -> char* {
    char* p = ws + off;
    off = (off + bytes + 255) & ~(size_t)255;
    return p;
  };
  __bf16* atf  = (__bf16*)take((size_t)2 * LMAX * NPH * sizeof(__bf16));
  __bf16* ati  = (__bf16*)take((size_t)2 * LMAX * NPH * sizeof(__bf16));
  __bf16* legA = (__bf16*)take((size_t)NMODES * LMAX * NTH * sizeof(__bf16));
  __bf16* legT = (__bf16*)take((size_t)NMODES * LMAX * NTH * sizeof(__bf16));
  float* Fre   = (float*)take((size_t)NTH * LMAX * CIN * sizeof(float));
  float* Fim   = (float*)take((size_t)NTH * LMAX * CIN * sizeof(float));
  float* flmRe = (float*)take((size_t)LMAX * NMODES * CIN * sizeof(float));
  float* flmIm = (float*)take((size_t)LMAX * NMODES * CIN * sizeof(float));
  float* yRe   = (float*)take((size_t)LMAX * NMODES * COUT * sizeof(float));
  float* yIm   = (float*)take((size_t)LMAX * NMODES * COUT * sizeof(float));
  float* GRe   = (float*)take((size_t)NMODES * NTH * COUT * sizeof(float));
  float* GIm   = (float*)take((size_t)NMODES * NTH * COUT * sizeof(float));

  k_tables<<<dim3(1024), dim3(256), 0, stream>>>(atf, ati);
  k_prep<<<dim3(LMAX * NMODES), dim3(256), 0, stream>>>(leg, qw, legA, legT);
  k_dft<<<dim3(NTH), dim3(256), 0, stream>>>(x, atf, Fre, Fim);
  k_leg_fwd<<<dim3(NMODES), dim3(256), 0, stream>>>(legA, Fre, Fim, flmRe, flmIm);
  k_wapply<<<dim3(LMAX * NMODES / 8), dim3(256), 0, stream>>>(flmRe, flmIm, Wr, Wi, yRe, yIm);
  k_leg_inv<<<dim3(NMODES), dim3(256), 0, stream>>>(legT, yRe, yIm, GRe, GIm);
  k_idft<<<dim3(NTH), dim3(256), 0, stream>>>(GRe, GIm, ati, out);
}
